// MambaBlock_15504831939006
// MI455X (gfx1250) — compile-verified, matrix operand order checked by reading, not run on hardware
//
#include <hip/hip_runtime.h>
#include <hip/hip_bf16.h>
#include <stdint.h>

// ---------------- problem constants ----------------
constexpr int B_SZ    = 4;
constexpr int SEQ     = 2048;
constexpr int D_MODEL = 1024;
constexpr int D_INNER = 2048;
constexpr int D_STATE = 16;
constexpr int DT_RANK = 64;
constexpr int M_ROWS  = B_SZ * SEQ;          // 8192
constexpr int N_XZ    = 2 * D_INNER;         // 4096
constexpr int N_XDBL  = DT_RANK + 2*D_STATE; // 96

typedef __bf16 bf16;
typedef __attribute__((ext_vector_type(16))) __bf16 v16bf;
typedef __attribute__((ext_vector_type(8)))  __bf16 v8bf;
typedef __attribute__((ext_vector_type(8)))  float  v8f;

// ---------------- TDM (Tensor Data Mover) descriptor, ISA cdna5 ch.8 ----------------
#if __has_builtin(__builtin_amdgcn_tensor_load_to_lds)
#define USE_TDM 1
typedef __attribute__((ext_vector_type(4))) unsigned int tdm_u32x4;
typedef __attribute__((ext_vector_type(8))) int          tdm_i32x8;
typedef __attribute__((ext_vector_type(4))) int          tdm_i32x4;

// Load a [128 rows x 64 elems] bf16 tile (row stride = K elems) from gptr into
// LDS offset 0. Rows >= nValid read OOB -> zero-filled (used by the N=96 GEMM).
__device__ inline void tdm_load_b_tile(const bf16* gptr, int K, int nValid) {
    unsigned long long ga = (unsigned long long)gptr;
    tdm_u32x4 g0;
    g0[0] = 1u;                                            // count=1, user mode
    g0[1] = 0u;                                            // lds_addr = 0 (sB is the only LDS object)
    g0[2] = (unsigned)(ga & 0xffffffffu);                  // global_addr lo
    g0[3] = (unsigned)((ga >> 32) & 0x01ffffffu) | (2u << 30); // global_addr hi, type=2("image")
    unsigned td0 = (unsigned)K;        // tensor_dim0 (elements along K)
    unsigned td1 = (unsigned)nValid;   // tensor_dim1 (rows)
    unsigned long long s0 = (unsigned long long)K;  // tensor_dim0_stride (elements)
    tdm_i32x8 g1;
    g1[0] = (int)(1u << 16);                               // workgroup_mask=0, data_size=1 (2B)
    g1[1] = (int)((td0 & 0xffffu) << 16);                  // barrier_addr=0 | tensor_dim0[15:0]
    g1[2] = (int)((td0 >> 16) | ((td1 & 0xffffu) << 16));  // tensor_dim0[31:16] | tensor_dim1[15:0]
    g1[3] = (int)((td1 >> 16) | (64u << 16));              // tensor_dim1[31:16] | tile_dim0=64
    g1[4] = (int)128u;                                     // tile_dim1=128, tile_dim2=0
    g1[5] = (int)(s0 & 0xffffffffu);                       // tensor_dim0_stride[31:0]
    g1[6] = (int)((s0 >> 32) & 0xffffu);                   // stride[47:32] | tensor_dim1_stride lo16=0
    g1[7] = 0;
    tdm_i32x4 gz4 = {0, 0, 0, 0};                          // groups 2/3 unused (2D tensor)
    tdm_i32x8 gz8 = {0, 0, 0, 0, 0, 0, 0, 0};
    // 6-arg form (this toolchain): (g0, g1, g2, g3, ext, cpol)
    __builtin_amdgcn_tensor_load_to_lds(g0, g1, gz4, gz4, gz8, 0);
}
#else
#define USE_TDM 0
#endif

// ---------------- helpers ----------------
__device__ inline v16bf pack16(const bf16* p) {
    // WMMA 16-bit fragment: per lane, two contiguous 8-element (16B) K-chunks
    // at +0 and +16 elements (CDNA5 ISA A-matrix 16x32 layout).
    v8bf lo = *(const v8bf*)p;
    v8bf hi = *(const v8bf*)(p + 16);
    v16bf r;
#pragma unroll
    for (int i = 0; i < 8; ++i) { r[i] = lo[i]; r[i + 8] = hi[i]; }
    return r;
}

// ---------------- generic bf16 WMMA GEMM: C[M,N] = A[M,K] * W[N,K]^T ----------------
// block: 256 thr (8 waves); block tile 128x128; wave tile 64x32 (4x2 WMMA frags).
// B tile (128 x 64 bf16) staged in LDS via TDM per 64-wide K chunk.
// Requires: M % 128 == 0, K % 64 == 0. N handled with store guard / OOB-zero loads.
__global__ __launch_bounds__(256)
void k_gemm_bf16(float* __restrict__ C, const bf16* __restrict__ A,
                 const bf16* __restrict__ W, int M, int N, int K, int ldc)
{
    __shared__ bf16 sB[128 * 64];   // MUST stay the only LDS object (TDM lds_addr=0)

    const int tid    = threadIdx.x;
    const int lane   = tid & 31;
    const int wid    = tid >> 5;     // 0..7
    const int wm     = wid >> 2;     // 0..1  (M dir)
    const int wn     = wid & 3;      // 0..3  (N dir)
    const int mBlock = blockIdx.x * 128;
    const int nBlock = blockIdx.y * 128;
    const int mlo    = lane & 15;
    const int koff   = (lane >> 4) << 3;   // 0 or 8 elements

    v8f acc[4][2];
#pragma unroll
    for (int mi = 0; mi < 4; ++mi)
#pragma unroll
        for (int ni = 0; ni < 2; ++ni) acc[mi][ni] = (v8f)0.f;

    for (int kk = 0; kk < K; kk += 64) {
        // ---- stage B tile [nBlock..nBlock+127] x [kk..kk+63] into LDS ----
#if USE_TDM
        if (wid == 0) {
            tdm_load_b_tile(W + (size_t)nBlock * K + kk, K, N - nBlock);
            __builtin_amdgcn_s_wait_tensorcnt(0);
        }
        // The TDM write to LDS is invisible to the compiler's alias analysis;
        // force it to treat sB as modified so the ds_load fragment reads below
        // are not folded/hoisted as loads of never-written memory.
        asm volatile("" : : "r"(&sB[0]) : "memory");
#else
        for (int i = tid; i < 128 * 16; i += 256) {       // 4 bf16 (8B) per iter
            int r  = i >> 4;
            int c4 = (i & 15) << 2;
            int gn = nBlock + r; if (gn >= N) gn = N - 1; // clamp (masked at store)
            *(uint2*)&sB[r * 64 + c4] =
                *(const uint2*)&W[(size_t)gn * K + kk + c4];
        }
#endif
        __syncthreads();

        if (kk + 64 < K)  // warm L2 for the next A chunk
            __builtin_prefetch(&A[(size_t)(mBlock + wm * 64 + mlo) * K + kk + 64], 0, 1);

#pragma unroll
        for (int ks = 0; ks < 64; ks += 32) {
            const int kg = kk + ks;
            v16bf bfrag[2];
#pragma unroll
            for (int ni = 0; ni < 2; ++ni)
                bfrag[ni] = pack16(&sB[(wn * 32 + ni * 16 + mlo) * 64 + ks + koff]);
#pragma unroll
            for (int mi = 0; mi < 4; ++mi) {
                v16bf afrag =
                    pack16(&A[(size_t)(mBlock + wm * 64 + mi * 16 + mlo) * K + kg + koff]);
#pragma unroll
                for (int ni = 0; ni < 2; ++ni)
                    acc[mi][ni] = __builtin_amdgcn_wmma_f32_16x16x32_bf16(
                        false, afrag, false, bfrag[ni], (short)0, acc[mi][ni],
                        false, false);
            }
        }
        __syncthreads();   // protect sB reuse
    }

    // ---- store: C frag layout = 8 VGPRs, row = base + r + 8*(lane>=16), col = lane&15
    const int chi = (lane >> 4) * 8;
#pragma unroll
    for (int mi = 0; mi < 4; ++mi) {
#pragma unroll
        for (int ni = 0; ni < 2; ++ni) {
            int r0 = mBlock + wm * 64 + mi * 16 + chi;
            int cn = nBlock + wn * 32 + ni * 16 + mlo;
            if (cn < N) {
#pragma unroll
                for (int r = 0; r < 8; ++r)
                    C[(size_t)(r0 + r) * ldc + cn] = acc[mi][ni][r];
            }
        }
    }
}

// ---------------- elementwise: f32 -> bf16 cast ----------------
__global__ void k_cast_bf16(const float* __restrict__ src, bf16* __restrict__ dst, size_t n) {
    size_t i = (size_t)blockIdx.x * blockDim.x + threadIdx.x;
    size_t stride = (size_t)gridDim.x * blockDim.x;
    for (; i < n; i += stride) dst[i] = (bf16)src[i];
}

// dt_r = x_dbl[:, 0:64] -> bf16
__global__ void k_cast_dtr(const float* __restrict__ xdbl, bf16* __restrict__ dtr) {
    int idx = blockIdx.x * blockDim.x + threadIdx.x;   // < 8192*64
    int r = idx >> 6, c = idx & 63;
    dtr[idx] = (bf16)xdbl[(size_t)r * N_XDBL + c];
}

// ---------------- depthwise causal conv1d + bias + SiLU ----------------
// reads u = xz[:, 0:2048]; writes u_c in f32 and bf16
__global__ void k_conv_silu(const float* __restrict__ xz,
                            const float* __restrict__ cw, const float* __restrict__ cb,
                            float* __restrict__ uc, bf16* __restrict__ ucb)
{
    int idx = blockIdx.x * blockDim.x + threadIdx.x;   // < 8192*2048
    int d   = idx & (D_INNER - 1);
    int row = idx >> 11;
    int l   = row & (SEQ - 1);
    float acc = cb[d];
#pragma unroll
    for (int j = 0; j < 4; ++j)        // tap j steps back, weight cw[d][3-j]
        if (l - j >= 0)
            acc += cw[d * 4 + (3 - j)] * xz[(size_t)(row - j) * N_XZ + d];
    float s = acc / (1.f + __expf(-acc));    // SiLU
    uc[idx]  = s;
    ucb[idx] = (bf16)s;
}

// ---------------- selective scan (fused: softplus(dt)+bias, SSM, +u*D, *silu(z)) ----------------
// one thread per (b,d) channel, 16 states in registers, B/C staged in LDS per 32 steps
__global__ __launch_bounds__(256)
void k_scan(const float* __restrict__ xdbl, const float* __restrict__ dtraw,
            const float* __restrict__ uc,   const float* __restrict__ xz,
            const float* __restrict__ Alog, const float* __restrict__ Dv,
            const float* __restrict__ bdt,  bf16* __restrict__ ybf)
{
    __shared__ float sBC[32][32];   // [t][0..15]=B_t, [t][16..31]=C_t
    const int d = blockIdx.x * 256 + threadIdx.x;   // < D_INNER
    const int b = blockIdx.y;

    float Aa[D_STATE], h[D_STATE];
#pragma unroll
    for (int s = 0; s < D_STATE; ++s) {
        Aa[s] = -__expf(Alog[d * D_STATE + s]);
        h[s]  = 0.f;
    }
    const float Dd = Dv[d], bd = bdt[d];

    for (int lc = 0; lc < SEQ; lc += 32) {
        __syncthreads();
        for (int i = threadIdx.x; i < 32 * 32; i += 256) {
            int t = i >> 5, j = i & 31;
            sBC[t][j] = xdbl[(size_t)(b * SEQ + lc + t) * N_XDBL + DT_RANK + j];
        }
        __syncthreads();
#pragma unroll 4
        for (int t = 0; t < 32; ++t) {
            const size_t row = (size_t)(b * SEQ + lc + t);
            float dv = dtraw[row * D_INNER + d] + bd;
            dv = (dv > 20.f) ? dv : __logf(1.f + __expf(dv));  // softplus
            const float u  = uc[row * D_INNER + d];
            const float du = dv * u;
            float y = 0.f;
#pragma unroll
            for (int s = 0; s < D_STATE; ++s) {
                h[s] = __expf(dv * Aa[s]) * h[s] + du * sBC[t][s];
                y   += h[s] * sBC[t][16 + s];
            }
            const float z = xz[row * (size_t)N_XZ + D_INNER + d];
            y = (y + u * Dd) * (z / (1.f + __expf(-z)));       // +u*D, *silu(z)
            ybf[row * D_INNER + d] = (bf16)y;
        }
    }
}

// ---------------- host side ----------------
extern "C" void kernel_launch(void* const* d_in, const int* in_sizes, int n_in,
                              void* d_out, int out_size, void* d_ws, size_t ws_size,
                              hipStream_t stream)
{
    const float* x      = (const float*)d_in[0];
    const float* W_in   = (const float*)d_in[1];
    const float* conv_w = (const float*)d_in[2];
    const float* conv_b = (const float*)d_in[3];
    const float* W_x    = (const float*)d_in[4];
    const float* W_dt   = (const float*)d_in[5];
    const float* b_dt   = (const float*)d_in[6];
    const float* A_log  = (const float*)d_in[7];
    const float* Dv     = (const float*)d_in[8];
    const float* W_out  = (const float*)d_in[9];
    float* out = (float*)d_out;

    char* ws = (char*)d_ws;
    size_t off = 0;
    auto alloc = [&](size_t bytes) -> char* {
        char* p = ws + off;
        off = (off + bytes + 255) & ~(size_t)255;
        return p;
    };
    bf16*  x_bf   = (bf16*)alloc((size_t)M_ROWS * D_MODEL * 2);
    bf16*  Win_bf = (bf16*)alloc((size_t)N_XZ * D_MODEL * 2);
    bf16*  Wx_bf  = (bf16*)alloc((size_t)N_XDBL * D_INNER * 2);
    bf16*  Wdt_bf = (bf16*)alloc((size_t)D_INNER * DT_RANK * 2);
    bf16*  Wout_bf= (bf16*)alloc((size_t)D_MODEL * D_INNER * 2);
    float* xz     = (float*)alloc((size_t)M_ROWS * N_XZ * 4);
    float* uc     = (float*)alloc((size_t)M_ROWS * D_INNER * 4);
    bf16*  ucb    = (bf16*)alloc((size_t)M_ROWS * D_INNER * 2);
    float* xdbl   = (float*)alloc((size_t)M_ROWS * N_XDBL * 4);
    bf16*  dtr_bf = (bf16*)alloc((size_t)M_ROWS * DT_RANK * 2);
    float* dtraw  = (float*)alloc((size_t)M_ROWS * D_INNER * 4);
    bf16*  ybf    = (bf16*)alloc((size_t)M_ROWS * D_INNER * 2);
    (void)ws_size; (void)in_sizes; (void)n_in; (void)out_size;

    // 1) bf16 demotions
    k_cast_bf16<<<2048, 256, 0, stream>>>(x,     x_bf,   (size_t)M_ROWS * D_MODEL);
    k_cast_bf16<<<2048, 256, 0, stream>>>(W_in,  Win_bf, (size_t)N_XZ * D_MODEL);
    k_cast_bf16<<<512,  256, 0, stream>>>(W_x,   Wx_bf,  (size_t)N_XDBL * D_INNER);
    k_cast_bf16<<<512,  256, 0, stream>>>(W_dt,  Wdt_bf, (size_t)D_INNER * DT_RANK);
    k_cast_bf16<<<2048, 256, 0, stream>>>(W_out, Wout_bf,(size_t)D_MODEL * D_INNER);

    // 2) in_proj: xz = x * W_in^T    [8192 x 4096], K=1024
    k_gemm_bf16<<<dim3(M_ROWS / 128, N_XZ / 128), 256, 0, stream>>>(
        xz, x_bf, Win_bf, M_ROWS, N_XZ, D_MODEL, N_XZ);

    // 3) depthwise causal conv + bias + SiLU
    k_conv_silu<<<(M_ROWS * D_INNER) / 256, 256, 0, stream>>>(xz, conv_w, conv_b, uc, ucb);

    // 4) x_proj: x_dbl = u_c * W_x^T  [8192 x 96], K=2048 (one 128-wide N block, guarded)
    k_gemm_bf16<<<dim3(M_ROWS / 128, 1), 256, 0, stream>>>(
        xdbl, ucb, Wx_bf, M_ROWS, N_XDBL, D_INNER, N_XDBL);

    // 5) dt_proj: dt_raw = dt_r * W_dt^T  [8192 x 2048], K=64
    k_cast_dtr<<<(M_ROWS * DT_RANK) / 256, 256, 0, stream>>>(xdbl, dtr_bf);
    k_gemm_bf16<<<dim3(M_ROWS / 128, D_INNER / 128), 256, 0, stream>>>(
        dtraw, dtr_bf, Wdt_bf, M_ROWS, D_INNER, DT_RANK, D_INNER);

    // 6) fused selective scan + gating -> y (bf16)
    k_scan<<<dim3(D_INNER / 256, B_SZ), 256, 0, stream>>>(
        xdbl, dtraw, uc, xz, A_log, Dv, b_dt, ybf);

    // 7) out_proj: out = y * W_out^T  [8192 x 1024], K=2048
    k_gemm_bf16<<<dim3(M_ROWS / 128, D_MODEL / 128), 256, 0, stream>>>(
        out, ybf, Wout_bf, M_ROWS, D_MODEL, D_INNER, D_MODEL);
}
